// MultiHeadSelfAttention_188978561294
// MI455X (gfx1250) — matrix operand order repaired and checked by reading, not compiled, not run on hardware
//
#include <hip/hip_runtime.h>

#define D_MODEL 1024
#define NB 4
#define SEQ 2048
#define NH 16
#define DK 64
#define MTOT (NB * SEQ)
#define LN_EPS 1e-5f

typedef __attribute__((ext_vector_type(16))) __bf16 v16bf;
typedef __attribute__((ext_vector_type(8))) __bf16 v8bf;
typedef __attribute__((ext_vector_type(8))) float v8f;
typedef __attribute__((ext_vector_type(4))) int v4i;

typedef __attribute__((address_space(1))) v4i gas_v4i;
typedef __attribute__((address_space(3))) v4i las_v4i;

#if defined(__has_builtin)
#if __has_builtin(__builtin_amdgcn_global_load_async_to_lds_b128)
#define HAVE_ASYNC_LDS 1
#endif
#endif
#ifndef HAVE_ASYNC_LDS
#define HAVE_ASYNC_LDS 0
#endif

static __device__ __forceinline__ void wait_async_lds() {
#if defined(__has_builtin)
#if __has_builtin(__builtin_amdgcn_s_wait_asynccnt)
  __builtin_amdgcn_s_wait_asynccnt(0);
  return;
#endif
#endif
  asm volatile("s_wait_asynccnt 0x0" ::: "memory");
}

#if HAVE_ASYNC_LDS
static __device__ __forceinline__ void async_copy16(const __bf16* g, __bf16* l) {
  __builtin_amdgcn_global_load_async_to_lds_b128((gas_v4i*)g, (las_v4i*)l, 0, 0);
}
#endif

static __device__ __forceinline__ v8f v8f_zero() {
  v8f z = {0.f, 0.f, 0.f, 0.f, 0.f, 0.f, 0.f, 0.f};
  return z;
}

static __device__ __forceinline__ v8bf pack8(float4 a, float4 b) {
  v8bf p = {(__bf16)a.x, (__bf16)a.y, (__bf16)a.z, (__bf16)a.w,
            (__bf16)b.x, (__bf16)b.y, (__bf16)b.z, (__bf16)b.w};
  return p;
}

static __device__ __forceinline__ v8f wmma_bf16(v16bf a, v16bf b, v8f c) {
  return __builtin_amdgcn_wmma_f32_16x16x32_bf16(false, a, false, b, (short)0, c,
                                                 false, false);
}

static __device__ __forceinline__ v16bf frag_cat(v8bf lo, v8bf hi) {
  return __builtin_shufflevector(lo, hi, 0, 1, 2, 3, 4, 5, 6, 7, 8, 9, 10, 11,
                                 12, 13, 14, 15);
}

// A-matrix 16x32 bf16 fragment; per ISA 7.12.2 lane(m=l&15, half=l>>4) needs
// K = 8*half + [0..7] and K = 16 + 8*half + [0..7]: two aligned 16B LDS loads.
static __device__ __forceinline__ v16bf load_a_frag(const __bf16* tile, int ld,
                                                    int row0, int k0, int lane) {
  const int m = lane & 15, half = lane >> 4;
  const __bf16* rp = tile + (row0 + m) * ld + k0 + (half << 3);
  const v8bf lo = *(const v8bf*)rp;
  const v8bf hi = *(const v8bf*)(rp + 16);
  return frag_cat(lo, hi);
}

// B-matrix 32x16 bf16 fragment from tile stored [n][ld], k contiguous.
// Lane(n=l&15, kh=l>>4) needs K = 8*kh + [0..7] and K = 16 + 8*kh + [0..7]: two aligned 16B LDS loads.
static __device__ __forceinline__ v16bf load_b_frag_nt(const __bf16* tile, int ld,
                                                       int col0, int k0, int lane) {
  const int n = lane & 15, kh = lane >> 4;
  const __bf16* cp = tile + (col0 + n) * ld + k0 + (kh << 3);
  const v8bf lo = *(const v8bf*)cp;
  const v8bf hi = *(const v8bf*)(cp + 16);
  return frag_cat(lo, hi);
}

// ---------------------------------------------------------------------------
// Kernel 1: fused Q/K/V projection. y = x @ W^T + b, output bf16 [B,H,S,DK].
// grid (MTOT/64, D_MODEL/128, 3), block 256. Double-buffered LDS: next tile's
// global loads issue before current tile's WMMAs; one barrier per iteration.
// ---------------------------------------------------------------------------
__global__ void __launch_bounds__(256) qkv_proj_kernel(
    const float* __restrict__ x, const float* __restrict__ Wq,
    const float* __restrict__ bq, const float* __restrict__ Wk,
    const float* __restrict__ bk, const float* __restrict__ Wv,
    const float* __restrict__ bv, __bf16* __restrict__ Qo,
    __bf16* __restrict__ Ko, __bf16* __restrict__ Vo) {
  __shared__ __bf16 As[2][64][40];
  __shared__ __bf16 Bs[2][128][40];

  const int tid = threadIdx.x;
  const int lane = tid & 31;
  const int wid = tid >> 5;
  const int wm = wid & 3;
  const int wn = wid >> 2;
  const int m0 = blockIdx.x * 64;
  const int n0 = blockIdx.y * 128;

  const float* W;
  const float* bias;
  __bf16* out;
  if (blockIdx.z == 0) {
    W = Wq; bias = bq; out = Qo;
  } else if (blockIdx.z == 1) {
    W = Wk; bias = bk; out = Ko;
  } else {
    W = Wv; bias = bv; out = Vo;
  }

  const int arow = tid >> 2, akk = (tid & 3) * 8;
  const int brow = tid >> 1, bkk = (tid & 1) * 16;
  const float* asrc = x + (size_t)(m0 + arow) * D_MODEL + akk;
  const float* bsrc = W + (size_t)(n0 + brow) * D_MODEL + bkk;

  v8f acc[4];
#pragma unroll
  for (int j = 0; j < 4; ++j) acc[j] = v8f_zero();

  float4 ra0, ra1, rb0, rb1, rb2, rb3;
  ra0 = ((const float4*)asrc)[0];
  ra1 = ((const float4*)asrc)[1];
  rb0 = ((const float4*)bsrc)[0];
  rb1 = ((const float4*)bsrc)[1];
  rb2 = ((const float4*)bsrc)[2];
  rb3 = ((const float4*)bsrc)[3];
  *(v8bf*)&As[0][arow][akk] = pack8(ra0, ra1);
  *(v8bf*)&Bs[0][brow][bkk] = pack8(rb0, rb1);
  *(v8bf*)&Bs[0][brow][bkk + 8] = pack8(rb2, rb3);
  __syncthreads();

  for (int kt = 0; kt < 32; ++kt) {
    const int cur = kt & 1;
    const bool more = (kt + 1) < 32;
    if (more) {  // issue next tile's global loads under current compute
      const float* an = asrc + (kt + 1) * 32;
      const float* bn = bsrc + (kt + 1) * 32;
      ra0 = ((const float4*)an)[0];
      ra1 = ((const float4*)an)[1];
      rb0 = ((const float4*)bn)[0];
      rb1 = ((const float4*)bn)[1];
      rb2 = ((const float4*)bn)[2];
      rb3 = ((const float4*)bn)[3];
      if (kt + 2 < 32) {
        __builtin_prefetch(asrc + (kt + 2) * 32, 0, 0);
        __builtin_prefetch(bsrc + (kt + 2) * 32, 0, 0);
      }
    }
    const v16bf a = load_a_frag(&As[cur][0][0], 40, wm * 16, 0, lane);
    v16bf bf[4];
#pragma unroll
    for (int j = 0; j < 4; ++j)
      bf[j] = load_b_frag_nt(&Bs[cur][0][0], 40, wn * 64 + j * 16, 0, lane);
#pragma unroll
    for (int j = 0; j < 4; ++j) acc[j] = wmma_bf16(a, bf[j], acc[j]);
    if (more) {
      const int nxt = cur ^ 1;
      *(v8bf*)&As[nxt][arow][akk] = pack8(ra0, ra1);
      *(v8bf*)&Bs[nxt][brow][bkk] = pack8(rb0, rb1);
      *(v8bf*)&Bs[nxt][brow][bkk + 8] = pack8(rb2, rb3);
    }
    __syncthreads();
  }

  const int half = lane >> 4;
  const int cn = lane & 15;
#pragma unroll
  for (int j = 0; j < 4; ++j) {
    const int gn = n0 + wn * 64 + j * 16 + cn;
    const float bval = bias[gn];
    const int hh = gn >> 6, dd = gn & 63;
#pragma unroll
    for (int r = 0; r < 8; ++r) {
      const int gm = m0 + wm * 16 + r + (half << 3);
      const int bb = gm >> 11;  // / SEQ
      const int ss = gm & (SEQ - 1);
      out[((size_t)(bb * NH + hh) * SEQ + ss) * DK + dd] = (__bf16)(acc[j][r] + bval);
    }
  }
}

// ---------------------------------------------------------------------------
// Kernel 2: flash attention. grid (SEQ/128, B*H), block 256 (8 waves, 16 q
// rows each). Double-buffered K/V tiles: next K tile async-copies to LDS and
// next V tile loads to registers while current tile computes; one barrier per
// key-tile. P staging is per-wave LDS (same-wave DS ordering, no barrier).
// ---------------------------------------------------------------------------
__global__ void __launch_bounds__(256) attention_kernel(
    const __bf16* __restrict__ Q, const __bf16* __restrict__ K,
    const __bf16* __restrict__ V, __bf16* __restrict__ ctx) {
  __shared__ __bf16 Qs[128][72];
  __shared__ __bf16 Ks[2][64][72];
  __shared__ __bf16 Vt[2][64][72];  // [dk][key]
  __shared__ __bf16 Ps[8][16][72];

  const int tid = threadIdx.x, lane = tid & 31, wid = tid >> 5;
  const int bh = blockIdx.y;  // b*NH + h
  const int q0 = blockIdx.x * 128;
  const size_t base = (size_t)bh * SEQ * DK;
  const int half = lane >> 4, cn = lane & 15;
  const int vrow = tid >> 2, vd0 = (tid & 3) * 16;  // V staging assignment

  v8bf rv0, rv1;  // in-flight V tile registers

  auto issue_ktile = [&](int kb, int buf) {
    const __bf16* kg = K + base + (size_t)kb * DK;
#pragma unroll
    for (int i = 0; i < 2; ++i) {
      const int c = tid * 2 + i;
      const int row = c >> 3, off = (c & 7) * 8;
#if HAVE_ASYNC_LDS
      async_copy16(kg + row * DK + off, &Ks[buf][row][off]);
#else
      *(v8bf*)&Ks[buf][row][off] = *(const v8bf*)(kg + row * DK + off);
#endif
    }
  };
  auto load_vregs = [&](int kb) {
    const v8bf* vsrc = (const v8bf*)(V + base + (size_t)(kb + vrow) * DK + vd0);
    rv0 = vsrc[0];
    rv1 = vsrc[1];
  };
  auto store_vt = [&](int buf) {
#pragma unroll
    for (int i = 0; i < 8; ++i) Vt[buf][vd0 + i][vrow] = rv0[i];
#pragma unroll
    for (int i = 0; i < 8; ++i) Vt[buf][vd0 + 8 + i][vrow] = rv1[i];
  };

  {  // stage Q tile 128x64 bf16 : 1024 16B chunks, 4 per thread
    const __bf16* qg = Q + base + (size_t)q0 * DK;
#pragma unroll
    for (int i = 0; i < 4; ++i) {
      const int c = tid * 4 + i;
      const int row = c >> 3, off = (c & 7) * 8;
#if HAVE_ASYNC_LDS
      async_copy16(qg + row * DK + off, &Qs[row][off]);
#else
      *(v8bf*)&Qs[row][off] = *(const v8bf*)(qg + row * DK + off);
#endif
    }
  }
  issue_ktile(0, 0);
  load_vregs(0);
#if HAVE_ASYNC_LDS
  wait_async_lds();
#endif
  store_vt(0);
  __syncthreads();

  v16bf aq[2];
#pragma unroll
  for (int kt = 0; kt < 2; ++kt)
    aq[kt] = load_a_frag(&Qs[0][0], 72, wid * 16, kt * 32, lane);

  v8f o[4];
  float m_i[8], l_i[8];
#pragma unroll
  for (int j = 0; j < 4; ++j) o[j] = v8f_zero();
#pragma unroll
  for (int r = 0; r < 8; ++r) { m_i[r] = -1e30f; l_i[r] = 0.0f; }

  for (int it = 0; it < SEQ / 64; ++it) {
    const int cur = it & 1, nxt = cur ^ 1;
    const bool more = (it + 1) < SEQ / 64;
    if (more) {  // overlap next tile's data movement with current compute
      issue_ktile((it + 1) * 64, nxt);
      load_vregs((it + 1) * 64);
    }

    // scores S = Q K^T * (1/sqrt(dk))
    v8f s[4];
#pragma unroll
    for (int j = 0; j < 4; ++j) s[j] = v8f_zero();
#pragma unroll
    for (int ktk = 0; ktk < 2; ++ktk) {
      v16bf bf[4];
#pragma unroll
      for (int j = 0; j < 4; ++j)
        bf[j] = load_b_frag_nt(&Ks[cur][0][0], 72, j * 16, ktk * 32, lane);
#pragma unroll
      for (int j = 0; j < 4; ++j) s[j] = wmma_bf16(aq[ktk], bf[j], s[j]);
    }
#pragma unroll
    for (int j = 0; j < 4; ++j)
#pragma unroll
      for (int r = 0; r < 8; ++r) s[j][r] *= 0.125f;

    // online softmax per row (row m = r + 8*half lives in 16 lanes of a half)
#pragma unroll
    for (int r = 0; r < 8; ++r) {
      float tmax = -1e30f;
#pragma unroll
      for (int j = 0; j < 4; ++j) tmax = fmaxf(tmax, s[j][r]);
      tmax = fmaxf(tmax, __shfl_xor(tmax, 1, 32));
      tmax = fmaxf(tmax, __shfl_xor(tmax, 2, 32));
      tmax = fmaxf(tmax, __shfl_xor(tmax, 4, 32));
      tmax = fmaxf(tmax, __shfl_xor(tmax, 8, 32));
      const float mnew = fmaxf(m_i[r], tmax);
      const float alpha = __expf(m_i[r] - mnew);
      m_i[r] = mnew;
      float rsum = 0.0f;
#pragma unroll
      for (int j = 0; j < 4; ++j) {
        const float p = __expf(s[j][r] - mnew);
        Ps[wid][r + (half << 3)][j * 16 + cn] = (__bf16)p;
        rsum += p;
      }
      rsum += __shfl_xor(rsum, 1, 32);
      rsum += __shfl_xor(rsum, 2, 32);
      rsum += __shfl_xor(rsum, 4, 32);
      rsum += __shfl_xor(rsum, 8, 32);
      l_i[r] = l_i[r] * alpha + rsum;
#pragma unroll
      for (int j = 0; j < 4; ++j) o[j][r] *= alpha;
    }

    // O += P V  (Ps is per-wave; same-wave DS ordering makes this safe)
#pragma unroll
    for (int ktk = 0; ktk < 2; ++ktk) {
      const v16bf ap = load_a_frag(&Ps[wid][0][0], 72, 0, ktk * 32, lane);
      v16bf bf[4];
#pragma unroll
      for (int j = 0; j < 4; ++j)
        bf[j] = load_b_frag_nt(&Vt[cur][0][0], 72, j * 16, ktk * 32, lane);
#pragma unroll
      for (int j = 0; j < 4; ++j) o[j] = wmma_bf16(ap, bf[j], o[j]);
    }

    if (more) store_vt(nxt);
#if HAVE_ASYNC_LDS
    wait_async_lds();
#endif
    __syncthreads();
  }

  const int h = bh & (NH - 1);
  const int bidx = bh >> 4;
#pragma unroll
  for (int r = 0; r < 8; ++r) {
    const float inv = 1.0f / l_i[r];
    const int srow = q0 + wid * 16 + r + (half << 3);
    __bf16* dst = ctx + ((size_t)(bidx * SEQ + srow)) * D_MODEL + h * DK;
#pragma unroll
    for (int j = 0; j < 4; ++j) dst[j * 16 + cn] = (__bf16)(o[j][r] * inv);
  }
}

// ---------------------------------------------------------------------------
// Kernel 3: out projection + residual + LayerNorm, fused. Each block owns 16
// full rows (1024 cols, two 512-col passes). grid MTOT/16, block 256.
// ---------------------------------------------------------------------------
__global__ void __launch_bounds__(256) oproj_ln_kernel(
    const __bf16* __restrict__ ctx, const float* __restrict__ Wo,
    const float* __restrict__ bo, const float* __restrict__ x,
    const float* __restrict__ gamma, const float* __restrict__ beta,
    float* __restrict__ out) {
  __shared__ __bf16 As[16][40];
  __shared__ __bf16 Bs[512][40];
  __shared__ float red_s[16][8];
  __shared__ float red_q[16][8];
  __shared__ float mu_s[16], rs_s[16];

  const int tid = threadIdx.x, lane = tid & 31, wid = tid >> 5;
  const int m0 = blockIdx.x * 16;
  const int half = lane >> 4, cn = lane & 15;

  v8f acc[8];
#pragma unroll
  for (int a = 0; a < 8; ++a) acc[a] = v8f_zero();

  for (int np = 0; np < 2; ++np) {
    const int brow = tid * 2;
    const float* bsrc0 = Wo + (size_t)(np * 512 + brow) * D_MODEL;
    const float* bsrc1 = Wo + (size_t)(np * 512 + brow + 1) * D_MODEL;
    for (int k0 = 0; k0 < D_MODEL; k0 += 32) {
      __syncthreads();
      if (k0 + 32 < D_MODEL) {
        __builtin_prefetch(bsrc0 + k0 + 32, 0, 0);
        __builtin_prefetch(bsrc1 + k0 + 32, 0, 0);
      }
      {  // A tile 16x32 bf16: 64 16B chunks, threads 0..63
        if (tid < 64) {
          const int row = tid >> 2, off = (tid & 3) * 8;
          const __bf16* src = ctx + (size_t)(m0 + row) * D_MODEL + k0 + off;
#if HAVE_ASYNC_LDS
          async_copy16(src, &As[row][off]);
#else
          *(v8bf*)&As[row][off] = *(const v8bf*)src;
#endif
        }
      }
      {  // B tile 512x32 f32 -> bf16: 2 rows x 32 per thread
        const float4* s0 = (const float4*)(bsrc0 + k0);
        const float4* s1 = (const float4*)(bsrc1 + k0);
        *(v8bf*)&Bs[brow][0] = pack8(s0[0], s0[1]);
        *(v8bf*)&Bs[brow][8] = pack8(s0[2], s0[3]);
        *(v8bf*)&Bs[brow][16] = pack8(s0[4], s0[5]);
        *(v8bf*)&Bs[brow][24] = pack8(s0[6], s0[7]);
        *(v8bf*)&Bs[brow + 1][0] = pack8(s1[0], s1[1]);
        *(v8bf*)&Bs[brow + 1][8] = pack8(s1[2], s1[3]);
        *(v8bf*)&Bs[brow + 1][16] = pack8(s1[4], s1[5]);
        *(v8bf*)&Bs[brow + 1][24] = pack8(s1[6], s1[7]);
      }
#if HAVE_ASYNC_LDS
      wait_async_lds();
#endif
      __syncthreads();
      const v16bf a = load_a_frag(&As[0][0], 40, 0, 0, lane);
      v16bf bf[4];
#pragma unroll
      for (int t = 0; t < 4; ++t)
        bf[t] = load_b_frag_nt(&Bs[0][0], 40, wid * 64 + t * 16, 0, lane);
#pragma unroll
      for (int t = 0; t < 4; ++t)
        acc[np * 4 + t] = wmma_bf16(a, bf[t], acc[np * 4 + t]);
    }
  }

  // epilogue: h = acc + b_o + x ; LayerNorm over D per row
  float psum[8], psq[8];
#pragma unroll
  for (int r = 0; r < 8; ++r) { psum[r] = 0.f; psq[r] = 0.f; }
#pragma unroll
  for (int a = 0; a < 8; ++a) {
    const int np = a >> 2, t = a & 3;
    const int col = np * 512 + wid * 64 + t * 16 + cn;
    const float bval = bo[col];
#pragma unroll
    for (int r = 0; r < 8; ++r) {
      const int row = m0 + r + (half << 3);
      const float h = acc[a][r] + bval + x[(size_t)row * D_MODEL + col];
      acc[a][r] = h;
      psum[r] += h;
      psq[r] += h * h;
    }
  }
#pragma unroll
  for (int r = 0; r < 8; ++r) {
#pragma unroll
    for (int m = 1; m <= 8; m <<= 1) {
      psum[r] += __shfl_xor(psum[r], m, 32);
      psq[r] += __shfl_xor(psq[r], m, 32);
    }
  }
  if (cn == 0) {
#pragma unroll
    for (int r = 0; r < 8; ++r) {
      red_s[r + (half << 3)][wid] = psum[r];
      red_q[r + (half << 3)][wid] = psq[r];
    }
  }
  __syncthreads();
  if (tid < 16) {
    float s = 0.f, q = 0.f;
#pragma unroll
    for (int w = 0; w < 8; ++w) { s += red_s[tid][w]; q += red_q[tid][w]; }
    const float mu = s * (1.0f / D_MODEL);
    const float var = q * (1.0f / D_MODEL) - mu * mu;
    mu_s[tid] = mu;
    rs_s[tid] = rsqrtf(var + LN_EPS);
  }
  __syncthreads();
#pragma unroll
  for (int a = 0; a < 8; ++a) {
    const int np = a >> 2, t = a & 3;
    const int col = np * 512 + wid * 64 + t * 16 + cn;
    const float g = gamma[col], bt = beta[col];
#pragma unroll
    for (int r = 0; r < 8; ++r) {
      const int rloc = r + (half << 3);
      out[(size_t)(m0 + rloc) * D_MODEL + col] =
          (acc[a][r] - mu_s[rloc]) * rs_s[rloc] * g + bt;
    }
  }
}

// ---------------------------------------------------------------------------
extern "C" void kernel_launch(void* const* d_in, const int* in_sizes, int n_in,
                              void* d_out, int out_size, void* d_ws, size_t ws_size,
                              hipStream_t stream) {
  const float* x  = (const float*)d_in[0];
  const float* Wq = (const float*)d_in[1];
  const float* bq = (const float*)d_in[2];
  const float* Wk = (const float*)d_in[3];
  const float* bk = (const float*)d_in[4];
  const float* Wv = (const float*)d_in[5];
  const float* bv = (const float*)d_in[6];
  const float* Wo = (const float*)d_in[7];
  const float* bo = (const float*)d_in[8];
  const float* ga = (const float*)d_in[9];
  const float* be = (const float*)d_in[10];
  float* out = (float*)d_out;

  const size_t elems = (size_t)MTOT * D_MODEL;
  __bf16* Qb = (__bf16*)d_ws;
  __bf16* Kb = Qb + elems;
  __bf16* Vb = Kb + elems;
  __bf16* Cb = Vb + elems;

  dim3 g1(MTOT / 64, D_MODEL / 128, 3);
  qkv_proj_kernel<<<g1, 256, 0, stream>>>(x, Wq, bq, Wk, bk, Wv, bv, Qb, Kb, Vb);

  dim3 g2(SEQ / 128, NB * NH);
  attention_kernel<<<g2, 256, 0, stream>>>(Qb, Kb, Vb, Cb);

  oproj_ln_kernel<<<MTOT / 16, 256, 0, stream>>>(Cb, Wo, bo, x, ga, be, out);
}
